// PolicyValueNet_25752623907171
// MI455X (gfx1250) — matrix operand order repaired and checked by reading, not compiled, not run on hardware
//
#include <hip/hip_runtime.h>
#include <hip/hip_bf16.h>
#include <stdint.h>

// ---------------------------------------------------------------------------
// CDNA5 / gfx1250 implementation of the AttentiveFP policy-value net.
// Dense matmuls run on v_wmma_f32_16x16x32_bf16 (bf16 in, f32 accumulate).
// ---------------------------------------------------------------------------

typedef __bf16 bf16_t;
typedef __attribute__((ext_vector_type(16))) __bf16 v16bf;
typedef __attribute__((ext_vector_type(8)))  __bf16 v8bf;
typedef __attribute__((ext_vector_type(8)))  float  v8f;

#define LEAKY 0.01f
#define V8F_ZERO (v8f){0.f,0.f,0.f,0.f,0.f,0.f,0.f,0.f}

static constexpr int Nn = 100000;   // atoms
static constexpr int Ee = 300000;   // directed bonds
static constexpr int Gg = 4096;     // graphs
static constexpr int Hh = 256;      // hidden
static constexpr int Aa = 1024;     // action dim

__device__ __forceinline__ v16bf cat8(v8bf lo, v8bf hi) {
  return __builtin_shufflevector(lo, hi, 0,1,2,3,4,5,6,7,8,9,10,11,12,13,14,15);
}
// two 16B b128 loads -> one 16-element bf16 fragment
__device__ __forceinline__ v16bf frag16(const bf16_t* p0, const bf16_t* p1) {
  return cat8(*(const v8bf*)p0, *(const v8bf*)p1);
}

__device__ __forceinline__ void atomicMaxF(float* addr, float val) {
  unsigned int* ua = (unsigned int*)addr;
  unsigned int old = *ua;
  while (__uint_as_float(old) < val) {
    unsigned int assumed = old;
    old = atomicCAS(ua, assumed, __float_as_uint(val));
    if (old == assumed) break;
  }
}

// ---------------------------------------------------------------------------
// Generic bf16 WMMA GEMM:  C[M x Nc] = act( A[M x K] * W[Nc x K]^T + bias )
// Layouts (A/B in this kernel's contiguous k order, same on both sides):
//   A frag: lane<16 holds row m, K=[k..k+15];
//           lane>=16 same row, K=[k+16..k+31].
//   B frag: lane<16 -> col n, K=[k..k+15]; lane>=16 -> col n, K=[k+16..k+31]
//           (contiguous rows of row-major W, since B = W^T).
//   D frag: element i -> row r0 + (lane<16 ? i : i+8), col c0+(lane&15).
// grid = (M/32, Nc/256), block = 256 (8 waves: 2 row-tiles x 4 col-tiles,
// each wave owns a 16x64 tile = 4 chained accumulators).
// act: 0=none, 1=leaky(0.01), 2=relu
// ---------------------------------------------------------------------------
__global__ __launch_bounds__(256) void gemm_bf16(
    const bf16_t* __restrict__ A, const bf16_t* __restrict__ W,
    float* __restrict__ C, const float* __restrict__ bias,
    int M, int K, int Nc, int act)
{
  const int lane = threadIdx.x & 31;
  const int wave = threadIdx.x >> 5;
  const int r0 = (blockIdx.x * 2 + (wave >> 2)) * 16;
  const int c0 = blockIdx.y * 256 + (wave & 3) * 64;
  int row = r0 + (lane & 15);
  if (row >= M) row = M - 1;
  const bf16_t* Arow = A + (size_t)row * K;
  const int kA = (lane >> 4) << 4;   // 0 / 16
  const int kB = (lane >> 4) << 4;   // 0 / 16
  const bf16_t* W0 = W + (size_t)(c0      + (lane & 15)) * K + kB;
  const bf16_t* W1 = W + (size_t)(c0 + 16 + (lane & 15)) * K + kB;
  const bf16_t* W2 = W + (size_t)(c0 + 32 + (lane & 15)) * K + kB;
  const bf16_t* W3 = W + (size_t)(c0 + 48 + (lane & 15)) * K + kB;

  v8f acc0 = V8F_ZERO, acc1 = V8F_ZERO, acc2 = V8F_ZERO, acc3 = V8F_ZERO;
  for (int k = 0; k < K; k += 32) {
    v16bf a  = frag16(Arow + k + kA, Arow + k + kA + 8);
    v16bf b0 = frag16(W0 + k, W0 + k + 8);
    v16bf b1 = frag16(W1 + k, W1 + k + 8);
    v16bf b2 = frag16(W2 + k, W2 + k + 8);
    v16bf b3 = frag16(W3 + k, W3 + k + 8);
    acc0 = __builtin_amdgcn_wmma_f32_16x16x32_bf16(false, a, false, b0, (short)0, acc0, false, false);
    acc1 = __builtin_amdgcn_wmma_f32_16x16x32_bf16(false, a, false, b1, (short)0, acc1, false, false);
    acc2 = __builtin_amdgcn_wmma_f32_16x16x32_bf16(false, a, false, b2, (short)0, acc2, false, false);
    acc3 = __builtin_amdgcn_wmma_f32_16x16x32_bf16(false, a, false, b3, (short)0, acc3, false, false);
  }
  v8f accs[4] = {acc0, acc1, acc2, acc3};
  const int mbase = r0 + ((lane < 16) ? 0 : 8);
#pragma unroll
  for (int t = 0; t < 4; ++t) {
    const int n = c0 + t * 16 + (lane & 15);
    const float bv = bias ? bias[n] : 0.0f;
#pragma unroll
    for (int i = 0; i < 8; ++i) {
      const int m = mbase + i;
      if (m < M) {
        float v = accs[t][i] + bv;
        if (act == 1)      v = (v > 0.0f) ? v : LEAKY * v;
        else if (act == 2) v = fmaxf(v, 0.0f);
        C[(size_t)m * Nc + n] = v;
      }
    }
  }
}

// ---------------------------------------------------------------------------
// Gate conv fused kernel:  for edge e,
//   xj = leaky( [xn[src[e]] (256) , e_emb[e] (96)] @ W1^T )   (256 wide)
//   alpha_pre[e] += xj . att_l
// The 300k x 256 xj tensor never touches memory: leaky + dot folded into the
// WMMA epilogue with a width-16 shfl_xor butterfly.
// grid = E/32, block 256. K = 352 fixed, Nc = 256 fixed.
// ---------------------------------------------------------------------------
__global__ __launch_bounds__(256) void gate_edge_gemm(
    const bf16_t* __restrict__ xnb, const bf16_t* __restrict__ eemb,
    const bf16_t* __restrict__ W, const float* __restrict__ att_l,
    const int* __restrict__ src, float* __restrict__ alpha_pre, int E)
{
  const int lane = threadIdx.x & 31;
  const int wave = threadIdx.x >> 5;
  const int r0 = (blockIdx.x * 2 + (wave >> 2)) * 16;
  const int c0 = (wave & 3) * 64;
  int e = r0 + (lane & 15);
  if (e >= E) e = E - 1;
  const bf16_t* Axn = xnb + (size_t)src[e] * 256;
  const bf16_t* Aed = eemb + (size_t)e * 96;
  const int kA = (lane >> 4) << 4;
  const int kB = (lane >> 4) << 4;
  const bf16_t* Wp[4];
#pragma unroll
  for (int t = 0; t < 4; ++t)
    Wp[t] = W + (size_t)(c0 + t * 16 + (lane & 15)) * 352 + kB;

  v8f acc[4] = {V8F_ZERO, V8F_ZERO, V8F_ZERO, V8F_ZERO};
  for (int k = 0; k < 352; k += 32) {
    const int ka = k + kA;                    // region switch aligns to 32
    const bf16_t* ap = (ka < 256) ? (Axn + ka) : (Aed + (ka - 256));
    v16bf a = frag16(ap, ap + 8);
#pragma unroll
    for (int t = 0; t < 4; ++t) {
      v16bf b = frag16(Wp[t] + k, Wp[t] + k + 8);
      acc[t] = __builtin_amdgcn_wmma_f32_16x16x32_bf16(false, a, false, b, (short)0, acc[t], false, false);
    }
  }
  float part[8];
#pragma unroll
  for (int i = 0; i < 8; ++i) part[i] = 0.0f;
#pragma unroll
  for (int t = 0; t < 4; ++t) {
    const float al = att_l[c0 + t * 16 + (lane & 15)];
#pragma unroll
    for (int i = 0; i < 8; ++i) {
      float v = acc[t][i];
      v = (v > 0.0f) ? v : LEAKY * v;
      part[i] += v * al;
    }
  }
#pragma unroll
  for (int i = 0; i < 8; ++i)
    for (int off = 8; off > 0; off >>= 1)
      part[i] += __shfl_xor(part[i], off, 16);    // reduce across 16-lane half
  if ((lane & 15) == 0) {
    const int base = r0 + ((lane < 16) ? 0 : 8);
#pragma unroll
    for (int i = 0; i < 8; ++i) {
      const int ee = base + i;
      if (ee < E) atomicAdd(&alpha_pre[ee], part[i]);
    }
  }
}

// --------------------------- small helper kernels ---------------------------
__global__ void cvt_f32_bf16(const float* __restrict__ in, bf16_t* __restrict__ out, size_t n) {
  size_t i = (size_t)blockIdx.x * blockDim.x + threadIdx.x;
  if (i < n) out[i] = (bf16_t)in[i];
}
__global__ void fill_f32(float* __restrict__ p, float v, size_t n) {
  size_t i = (size_t)blockIdx.x * blockDim.x + threadIdx.x;
  if (i < n) p[i] = v;
}

struct Tab9 { const float* t[9]; };
struct Tab3 { const float* t[3]; };
__global__ void embed_atoms(const int* __restrict__ x, Tab9 tabs, bf16_t* __restrict__ out, size_t total) {
  size_t i = (size_t)blockIdx.x * blockDim.x + threadIdx.x;
  if (i >= total) return;
  const int n = (int)(i / 288), col = (int)(i % 288);
  const int f = col >> 5, w = col & 31;
  out[i] = (bf16_t)tabs.t[f][(size_t)x[n * 9 + f] * 32 + w];
}
__global__ void embed_bonds(const int* __restrict__ ea, Tab3 tabs, bf16_t* __restrict__ out, size_t total) {
  size_t i = (size_t)blockIdx.x * blockDim.x + threadIdx.x;
  if (i >= total) return;
  const int e = (int)(i / 96), col = (int)(i % 96);
  const int f = col >> 5, w = col & 31;
  out[i] = (bf16_t)tabs.t[f][(size_t)ea[e * 3 + f] * 32 + w];
}

// per-row dual dot product over 256 cols: o1=X.v1, o2=X.v2 (o2 optional)
__global__ void rowdot2(const float* __restrict__ X, const float* __restrict__ v1,
                        const float* __restrict__ v2, float* __restrict__ o1,
                        float* __restrict__ o2, int rows) {
  const int row = blockIdx.x * 8 + (threadIdx.x >> 5);
  const int lane = threadIdx.x & 31;
  if (row >= rows) return;
  const float* xr = X + (size_t)row * 256;
  float s1 = 0.f, s2 = 0.f;
#pragma unroll
  for (int j = 0; j < 8; ++j) {
    const float xv = xr[lane + 32 * j];
    s1 += xv * v1[lane + 32 * j];
    if (v2) s2 += xv * v2[lane + 32 * j];
  }
  for (int off = 16; off > 0; off >>= 1) {
    s1 += __shfl_xor(s1, off, 32);
    s2 += __shfl_xor(s2, off, 32);
  }
  if (lane == 0) { o1[row] = s1; if (o2) o2[row] = s2; }
}

// a[e] = leaky( (srcIdx ? xs[srcIdx[e]] : xs[e]) + yd[dstIdx[e]] )
__global__ void edge_alpha(const float* __restrict__ xs, const int* __restrict__ srcIdx,
                           const float* __restrict__ yd, const int* __restrict__ dstIdx,
                           float* __restrict__ a, int E) {
  const int e = blockIdx.x * blockDim.x + threadIdx.x;
  if (e >= E) return;
  const float v = (srcIdx ? xs[srcIdx[e]] : xs[e]) + yd[dstIdx[e]];
  a[e] = (v > 0.0f) ? v : LEAKY * v;
}
__global__ void seg_max(const float* __restrict__ a, const int* __restrict__ dst,
                        float* __restrict__ amax, int E) {
  const int e = blockIdx.x * blockDim.x + threadIdx.x;
  if (e < E) atomicMaxF(&amax[dst[e]], a[e]);
}
__global__ void seg_expsum(const float* __restrict__ a, const int* __restrict__ dst,
                           const float* __restrict__ amax, float* __restrict__ eexp,
                           float* __restrict__ denom, int E) {
  const int e = blockIdx.x * blockDim.x + threadIdx.x;
  if (e >= E) return;
  const float ex = expf(a[e] - amax[dst[e]]);
  eexp[e] = ex;
  atomicAdd(&denom[dst[e]], ex);
}
// out[dst[e], :] += X[src[e], :] * eexp[e]/(denom[dst[e]]+1e-16)   (256 cols)
__global__ void scatter_msg(const float* __restrict__ X, const int* __restrict__ src,
                            const int* __restrict__ dst, const float* __restrict__ eexp,
                            const float* __restrict__ denom, float* __restrict__ out) {
  const int e = blockIdx.x;
  const int s = src ? src[e] : e;
  const int d = dst[e];
  const float c = eexp[e] / (denom[d] + 1e-16f);
  atomicAdd(&out[(size_t)d * 256 + threadIdx.x], X[(size_t)s * 256 + threadIdx.x] * c);
}
// out[seg[n], :] += X[n, :]
__global__ void scatter_rows(const float* __restrict__ X, const int* __restrict__ seg,
                             float* __restrict__ out) {
  const int n = blockIdx.x;
  atomicAdd(&out[(size_t)seg[n] * 256 + threadIdx.x], X[(size_t)n * 256 + threadIdx.x]);
}
// X = act(X + bias[col]); Xb = bf16(X).  act: 1=relu, 2=elu
__global__ void bias_act(float* __restrict__ X, const float* __restrict__ bias,
                         bf16_t* __restrict__ Xb, size_t count, int act) {
  size_t i = (size_t)blockIdx.x * blockDim.x + threadIdx.x;
  if (i >= count) return;
  float v = X[i] + (bias ? bias[i & 255] : 0.0f);
  if (act == 1) v = fmaxf(v, 0.0f);
  else          v = (v > 0.0f) ? v : (expf(v) - 1.0f);
  X[i] = v;
  Xb[i] = (bf16_t)v;
}
// GRU elementwise (gi from conv output, gh from hidden), fused trailing relu
__global__ void gru_elem(const float* __restrict__ gi, const float* __restrict__ gh,
                         const float* __restrict__ hprev, float* __restrict__ xout,
                         bf16_t* __restrict__ xoutb, int rows) {
  size_t i = (size_t)blockIdx.x * blockDim.x + threadIdx.x;
  if (i >= (size_t)rows * 256) return;
  const size_t row = i >> 8; const int c = (int)(i & 255);
  const size_t b = row * 768 + c;
  const float r = 1.0f / (1.0f + expf(-(gi[b]       + gh[b])));
  const float z = 1.0f / (1.0f + expf(-(gi[b + 256] + gh[b + 256])));
  const float n = tanhf(gi[b + 512] + r * gh[b + 512]);
  float o = (1.0f - z) * n + z * hprev[i];
  o = fmaxf(o, 0.0f);
  xout[i] = o;
  xoutb[i] = (bf16_t)o;
}
// value head: out[g] = sigmoid(tanh(V1[g,:].w + b))
__global__ void value_head(const float* __restrict__ V1, const float* __restrict__ w,
                           const float* __restrict__ b, float* __restrict__ out, int G) {
  const int g = blockIdx.x * blockDim.x + threadIdx.x;
  if (g >= G) return;
  float s = b[0];
  for (int c = 0; c < 256; ++c) s += V1[(size_t)g * 256 + c] * w[c];
  const float t = tanhf(s);
  out[g] = 1.0f / (1.0f + expf(-t));
}

// ----------------------------- host orchestration ---------------------------
static inline void launch_gemm(hipStream_t s, const bf16_t* A, const bf16_t* W, float* C,
                               const float* bias, int M, int K, int Nc, int act) {
  dim3 grid(M / 32, Nc / 256);
  gemm_bf16<<<grid, 256, 0, s>>>(A, W, C, bias, M, K, Nc, act);
}
static inline void cvt(hipStream_t s, const float* src, bf16_t* dst, size_t n) {
  cvt_f32_bf16<<<(unsigned)((n + 255) / 256), 256, 0, s>>>(src, dst, n);
}
static inline void fill(hipStream_t s, float* p, float v, size_t n) {
  fill_f32<<<(unsigned)((n + 255) / 256), 256, 0, s>>>(p, v, n);
}

extern "C" void kernel_launch(void* const* d_in, const int* in_sizes, int n_in,
                              void* d_out, int out_size, void* d_ws, size_t ws_size,
                              hipStream_t stream) {
  (void)in_sizes; (void)n_in; (void)out_size; (void)ws_size;
  // ---- inputs (setup_inputs() insertion order, depth-first) ----
  const int* x     = (const int*)d_in[0];
  const int* src   = (const int*)d_in[1];          // edge_index row 0
  const int* dstp  = (const int*)d_in[1] + Ee;     // edge_index row 1
  const int* eattr = (const int*)d_in[2];
  const int* batch = (const int*)d_in[3];
  const float* atomTab[9]; for (int i = 0; i < 9; ++i) atomTab[i] = (const float*)d_in[4 + i];
  const float* bondTab[3]; for (int i = 0; i < 3; ++i) bondTab[i] = (const float*)d_in[13 + i];
  const float* lin1_w = (const float*)d_in[16]; const float* lin1_b = (const float*)d_in[17];
  const float* gateW1 = (const float*)d_in[18]; const float* gateW2 = (const float*)d_in[19];
  const float* gAttL  = (const float*)d_in[20]; const float* gAttR  = (const float*)d_in[21];
  const float* gBias  = (const float*)d_in[22];
  // convs: [atom0, atom1, mol]
  const float* convW[3]   = {(const float*)d_in[27], (const float*)d_in[31], (const float*)d_in[43]};
  const float* convAS[3]  = {(const float*)d_in[28], (const float*)d_in[32], (const float*)d_in[44]};
  const float* convAD[3]  = {(const float*)d_in[29], (const float*)d_in[33], (const float*)d_in[45]};
  const float* convB[3]   = {(const float*)d_in[30], (const float*)d_in[34], (const float*)d_in[46]};
  // grus: [gru0, atom1, atom2, mol]
  const float* gruWih[4]  = {(const float*)d_in[23], (const float*)d_in[35], (const float*)d_in[39], (const float*)d_in[47]};
  const float* gruWhh[4]  = {(const float*)d_in[24], (const float*)d_in[36], (const float*)d_in[40], (const float*)d_in[48]};
  const float* gruBih[4]  = {(const float*)d_in[25], (const float*)d_in[37], (const float*)d_in[41], (const float*)d_in[49]};
  const float* gruBhh[4]  = {(const float*)d_in[26], (const float*)d_in[38], (const float*)d_in[42], (const float*)d_in[50]};
  const float* lin2_w = (const float*)d_in[51]; const float* lin2_b = (const float*)d_in[52];
  const float* pol0_w = (const float*)d_in[53]; const float* pol0_b = (const float*)d_in[54];
  const float* pol1_w = (const float*)d_in[55]; const float* pol1_b = (const float*)d_in[56];
  const float* val0_w = (const float*)d_in[57]; const float* val0_b = (const float*)d_in[58];
  const float* val1_w = (const float*)d_in[59]; const float* val1_b = (const float*)d_in[60];

  // ---- workspace bump allocator ----
  size_t off = 0;
  auto alloc = [&](size_t bytes) -> void* {
    void* p = (char*)d_ws + off;
    off += (bytes + 255) & ~(size_t)255;
    return p;
  };
  bf16_t* xembB = (bf16_t*)alloc((size_t)Nn * 288 * 2);
  bf16_t* eembB = (bf16_t*)alloc((size_t)Ee * 96 * 2);
  float*  xn    = (float*) alloc((size_t)Nn * 256 * 4);
  bf16_t* xnb   = (bf16_t*)alloc((size_t)Nn * 256 * 2);
  float*  hbuf  = (float*) alloc((size_t)Nn * 256 * 4);
  bf16_t* hbufB = (bf16_t*)alloc((size_t)Nn * 256 * 2);
  float*  hs    = (float*) alloc((size_t)Nn * 256 * 4);
  float*  gi    = (float*) alloc((size_t)Nn * 768 * 4);
  float*  gh    = (float*) alloc((size_t)Nn * 768 * 4);
  float*  aE    = (float*) alloc((size_t)Ee * 4);
  float*  eexp  = (float*) alloc((size_t)Ee * 4);
  float*  apre  = (float*) alloc((size_t)Ee * 4);
  float*  asv   = (float*) alloc((size_t)Nn * 4);
  float*  adv   = (float*) alloc((size_t)Nn * 4);
  float*  amax  = (float*) alloc((size_t)Nn * 4);
  float*  denom = (float*) alloc((size_t)Nn * 4);
  float*  outG  = (float*) alloc((size_t)Gg * 256 * 4);
  bf16_t* outGB = (bf16_t*)alloc((size_t)Gg * 256 * 2);
  float*  molH  = (float*) alloc((size_t)Gg * 256 * 4);
  bf16_t* molHB = (bf16_t*)alloc((size_t)Gg * 256 * 2);
  float*  hdG   = (float*) alloc((size_t)Gg * 256 * 4);
  float*  hg    = (float*) alloc((size_t)Gg * 256 * 4);
  bf16_t* hgB   = (bf16_t*)alloc((size_t)Gg * 256 * 2);
  float*  p1    = (float*) alloc((size_t)Gg * 256 * 4);
  bf16_t* p1B   = (bf16_t*)alloc((size_t)Gg * 256 * 2);
  float*  v1    = (float*) alloc((size_t)Gg * 256 * 4);
  // bf16 weight copies
  bf16_t* wLin1  = (bf16_t*)alloc((size_t)256 * 288 * 2);
  bf16_t* wGate1 = (bf16_t*)alloc((size_t)256 * 352 * 2);
  bf16_t* wGate2 = (bf16_t*)alloc((size_t)256 * 256 * 2);
  bf16_t* wConv[3]; for (int i = 0; i < 3; ++i) wConv[i] = (bf16_t*)alloc((size_t)256 * 256 * 2);
  bf16_t* wIh[4];   for (int i = 0; i < 4; ++i) wIh[i]   = (bf16_t*)alloc((size_t)768 * 256 * 2);
  bf16_t* wHh[4];   for (int i = 0; i < 4; ++i) wHh[i]   = (bf16_t*)alloc((size_t)768 * 256 * 2);
  bf16_t* wLin2  = (bf16_t*)alloc((size_t)256 * 256 * 2);
  bf16_t* wPol0  = (bf16_t*)alloc((size_t)256 * 256 * 2);
  bf16_t* wPol1  = (bf16_t*)alloc((size_t)1024 * 256 * 2);
  bf16_t* wVal0  = (bf16_t*)alloc((size_t)256 * 256 * 2);

  float* policy_out = (float*)d_out;               // G x 1024
  float* value_out  = (float*)d_out + (size_t)Gg * Aa;

  // ---- weight conversion f32 -> bf16 (L2-resident afterwards) ----
  cvt(stream, lin1_w, wLin1, (size_t)256 * 288);
  cvt(stream, gateW1, wGate1, (size_t)256 * 352);
  cvt(stream, gateW2, wGate2, (size_t)256 * 256);
  for (int i = 0; i < 3; ++i) cvt(stream, convW[i], wConv[i], (size_t)256 * 256);
  for (int i = 0; i < 4; ++i) { cvt(stream, gruWih[i], wIh[i], (size_t)768 * 256);
                                cvt(stream, gruWhh[i], wHh[i], (size_t)768 * 256); }
  cvt(stream, lin2_w, wLin2, (size_t)256 * 256);
  cvt(stream, pol0_w, wPol0, (size_t)256 * 256);
  cvt(stream, pol1_w, wPol1, (size_t)1024 * 256);
  cvt(stream, val0_w, wVal0, (size_t)256 * 256);

  // ---- embeddings ----
  { Tab9 t; for (int i = 0; i < 9; ++i) t.t[i] = atomTab[i];
    size_t tot = (size_t)Nn * 288;
    embed_atoms<<<(unsigned)((tot + 255) / 256), 256, 0, stream>>>(x, t, xembB, tot); }
  { Tab3 t; for (int i = 0; i < 3; ++i) t.t[i] = bondTab[i];
    size_t tot = (size_t)Ee * 96;
    embed_bonds<<<(unsigned)((tot + 255) / 256), 256, 0, stream>>>(eattr, t, eembB, tot); }

  // ---- lin1 + leaky ----
  launch_gemm(stream, xembB, wLin1, xn, lin1_b, Nn, 288, 256, /*leaky*/1);
  cvt(stream, xn, xnb, (size_t)Nn * 256);

  const unsigned eg = (Ee + 255) / 256;

  // ---- GATEConv ----
  rowdot2<<<(Nn + 7) / 8, 256, 0, stream>>>(xn, gAttR, nullptr, asv, nullptr, Nn);
  fill(stream, apre, 0.0f, Ee);
  gate_edge_gemm<<<Ee / 32, 256, 0, stream>>>(xnb, eembB, wGate1, gAttL, src, apre, Ee);
  edge_alpha<<<eg, 256, 0, stream>>>(apre, nullptr, asv, dstp, aE, Ee);
  fill(stream, amax, -3.0e38f, Nn); fill(stream, denom, 0.0f, Nn);
  seg_max<<<eg, 256, 0, stream>>>(aE, dstp, amax, Ee);
  seg_expsum<<<eg, 256, 0, stream>>>(aE, dstp, amax, eexp, denom, Ee);
  launch_gemm(stream, xnb, wGate2, hs, nullptr, Nn, 256, 256, 0);   // xn @ lin2_w^T
  fill(stream, hbuf, 0.0f, (size_t)Nn * 256);
  scatter_msg<<<Ee, 256, 0, stream>>>(hs, src, dstp, eexp, denom, hbuf);
  bias_act<<<(unsigned)(((size_t)Nn * 256 + 255) / 256), 256, 0, stream>>>(hbuf, gBias, hbufB, (size_t)Nn * 256, /*elu*/2);
  // gru0
  launch_gemm(stream, hbufB, wIh[0], gi, gruBih[0], Nn, 256, 768, 0);
  launch_gemm(stream, xnb,   wHh[0], gh, gruBhh[0], Nn, 256, 768, 0);
  gru_elem<<<(unsigned)(((size_t)Nn * 256 + 255) / 256), 256, 0, stream>>>(gi, gh, xn, xn, xnb, Nn);

  // ---- GATConv atom layers ----
  for (int L = 0; L < 2; ++L) {
    launch_gemm(stream, xnb, wConv[L], hs, nullptr, Nn, 256, 256, 0);
    rowdot2<<<(Nn + 7) / 8, 256, 0, stream>>>(hs, convAS[L], convAD[L], asv, adv, Nn);
    edge_alpha<<<eg, 256, 0, stream>>>(asv, src, adv, dstp, aE, Ee);
    fill(stream, amax, -3.0e38f, Nn); fill(stream, denom, 0.0f, Nn);
    seg_max<<<eg, 256, 0, stream>>>(aE, dstp, amax, Ee);
    seg_expsum<<<eg, 256, 0, stream>>>(aE, dstp, amax, eexp, denom, Ee);
    fill(stream, hbuf, 0.0f, (size_t)Nn * 256);
    scatter_msg<<<Ee, 256, 0, stream>>>(hs, src, dstp, eexp, denom, hbuf);
    bias_act<<<(unsigned)(((size_t)Nn * 256 + 255) / 256), 256, 0, stream>>>(hbuf, convB[L], hbufB, (size_t)Nn * 256, 2);
    launch_gemm(stream, hbufB, wIh[1 + L], gi, gruBih[1 + L], Nn, 256, 768, 0);
    launch_gemm(stream, xnb,   wHh[1 + L], gh, gruBhh[1 + L], Nn, 256, 768, 0);
    gru_elem<<<(unsigned)(((size_t)Nn * 256 + 255) / 256), 256, 0, stream>>>(gi, gh, xn, xn, xnb, Nn);
  }

  // ---- molecule phase ----
  const unsigned ng = (Nn + 255) / 256;
  fill(stream, outG, 0.0f, (size_t)Gg * 256);
  scatter_rows<<<Nn, 256, 0, stream>>>(xn, batch, outG);
  bias_act<<<(unsigned)(((size_t)Gg * 256 + 255) / 256), 256, 0, stream>>>(outG, nullptr, outGB, (size_t)Gg * 256, /*relu*/1);
  launch_gemm(stream, xnb, wConv[2], hs, nullptr, Nn, 256, 256, 0);  // hs_mol (const over T)
  rowdot2<<<(Nn + 7) / 8, 256, 0, stream>>>(hs, convAS[2], nullptr, asv, nullptr, Nn);
  for (int t = 0; t < 2; ++t) {
    launch_gemm(stream, outGB, wConv[2], hdG, nullptr, Gg, 256, 256, 0);
    rowdot2<<<(Gg + 7) / 8, 256, 0, stream>>>(hdG, convAD[2], nullptr, adv, nullptr, Gg);
    edge_alpha<<<ng, 256, 0, stream>>>(asv, nullptr, adv, batch, aE, Nn);
    fill(stream, amax, -3.0e38f, Gg); fill(stream, denom, 0.0f, Gg);
    seg_max<<<ng, 256, 0, stream>>>(aE, batch, amax, Nn);
    seg_expsum<<<ng, 256, 0, stream>>>(aE, batch, amax, eexp, denom, Nn);
    fill(stream, molH, 0.0f, (size_t)Gg * 256);
    scatter_msg<<<Nn, 256, 0, stream>>>(hs, nullptr, batch, eexp, denom, molH);
    bias_act<<<(unsigned)(((size_t)Gg * 256 + 255) / 256), 256, 0, stream>>>(molH, convB[2], molHB, (size_t)Gg * 256, 2);
    launch_gemm(stream, molHB, wIh[3], gi, gruBih[3], Gg, 256, 768, 0);
    launch_gemm(stream, outGB, wHh[3], gh, gruBhh[3], Gg, 256, 768, 0);
    gru_elem<<<(unsigned)(((size_t)Gg * 256 + 255) / 256), 256, 0, stream>>>(gi, gh, outG, outG, outGB, Gg);
  }

  // ---- heads ----
  launch_gemm(stream, outGB, wLin2, hg, lin2_b, Gg, 256, 256, 0);
  cvt(stream, hg, hgB, (size_t)Gg * 256);
  launch_gemm(stream, hgB, wPol0, p1, pol0_b, Gg, 256, 256, /*relu*/2);
  cvt(stream, p1, p1B, (size_t)Gg * 256);
  launch_gemm(stream, p1B, wPol1, policy_out, pol1_b, Gg, 256, 1024, 0);
  launch_gemm(stream, hgB, wVal0, v1, val0_b, Gg, 256, 256, /*relu*/2);
  value_head<<<(Gg + 255) / 256, 256, 0, stream>>>(v1, val1_w, val1_b, value_out, Gg);
}
